// PointerGeneratorSeq2Seq_71021579206750
// MI455X (gfx1250) — compile-verified, matrix operand order checked by reading, not run on hardware
//
#include <hip/hip_runtime.h>
#include <hip/hip_bf16.h>

// ---------------------------------------------------------------------------
// Pointer-generator seq2seq forward, CDNA5 (gfx1250, wave32, WMMA bf16).
// All GEMMs use V_WMMA_F32_16X16X32_BF16 (bf16 A/B, f32 accumulate).
// GEMM is 2x2 register-blocked; the N-remainder predicate is forced scalar
// (readfirstlane) and hoisted out of the K-loop so the hot loop is pure
// load+WMMA with no EXEC masking.
// ---------------------------------------------------------------------------

#define Bdim 32
#define Sdim 400
#define Hdim 512
#define Vdim 50000

typedef __bf16 bf16_t;
typedef __attribute__((ext_vector_type(16))) __bf16 v16bf;
typedef __attribute__((ext_vector_type(8)))  float  v8f;

__device__ __forceinline__ bf16_t f2bf(float f) {
    unsigned u = __builtin_bit_cast(unsigned, f);
    unsigned r = u + 0x7FFFu + ((u >> 16) & 1u);           // round-to-nearest-even
    return __builtin_bit_cast(bf16_t, (unsigned short)(r >> 16));
}
__device__ __forceinline__ float sigmoidf_(float x) { return 1.0f / (1.0f + expf(-x)); }

__device__ __forceinline__ v8f wmma_bf16(v16bf a, v16bf b, v8f c) {
    // (neg_a, A, neg_b, B, c_mod, C, reuse_a, reuse_b)
    return __builtin_amdgcn_wmma_f32_16x16x32_bf16(false, a, false, b, (short)0, c, false, false);
}

// ---------------------------------------------------------------------------
// Prep: f32 (K,N) row-major  ->  bf16 (N,K) row-major (transposed weight)
// ---------------------------------------------------------------------------
__global__ void k_convert_t(const float* __restrict__ in, bf16_t* __restrict__ out,
                            int K, int N) {
    size_t i = (size_t)blockIdx.x * blockDim.x + threadIdx.x;
    size_t total = (size_t)K * N;
    if (i >= total) return;
    int n = (int)(i / K);
    int k = (int)(i % K);
    out[i] = f2bf(in[(size_t)k * N + n]);
}

// Embedding gathers
__global__ void k_gather_x(const int* __restrict__ ids, const float* __restrict__ emb,
                           bf16_t* __restrict__ xbf, int rows) {
    size_t i = (size_t)blockIdx.x * blockDim.x + threadIdx.x;
    if (i >= (size_t)rows * Hdim) return;
    int r = (int)(i / Hdim), h = (int)(i % Hdim);
    xbf[i] = f2bf(emb[(size_t)ids[r] * Hdim + h]);
}
__global__ void k_gather_y(const int* __restrict__ ids, const float* __restrict__ emb,
                           float* __restrict__ yf, bf16_t* __restrict__ ybf) {
    int i = blockIdx.x * blockDim.x + threadIdx.x;
    if (i >= Bdim * Hdim) return;
    int b = i >> 9, h = i & (Hdim - 1);
    float v = emb[(size_t)ids[b] * Hdim + h];
    yf[i] = v; ybf[i] = f2bf(v);
}

// ---------------------------------------------------------------------------
// WMMA GEMM, 2x2 register blocking:
//   C(M,N) = A(M,K)bf16 @ Bt(N,K)bf16^T  [+bias] [relu]
// Each wave computes a 32x32 block (four 16x16 tiles), reusing each A/B
// fragment twice: 8 b128 loads -> 4 WMMAs per k-step.
// M multiple of 32, N multiple of 16, K multiple of 32. Odd last N-tile
// handled by a scalarized (readfirstlane) uniform branch -> 2x1 block.
// flags: bit0 = relu
// ---------------------------------------------------------------------------
__device__ __forceinline__ void store_tile(float* Cf, bf16_t* Cbf, const float* bias,
                                           int flags, int N, int m0, int n0, int lane,
                                           v8f acc) {
    int row = m0 + ((lane >> 4) << 3);
    int col = n0 + (lane & 15);
    float bb = bias ? bias[col] : 0.0f;
    for (int r = 0; r < 8; ++r) {
        float v = acc[r] + bb;
        if (flags & 1) v = fmaxf(v, 0.0f);
        size_t o = (size_t)(row + r) * N + col;
        if (Cf)  Cf[o]  = v;
        if (Cbf) Cbf[o] = f2bf(v);
    }
}

__global__ __launch_bounds__(256) void k_gemm(const bf16_t* __restrict__ A,
                                              const bf16_t* __restrict__ Bt,
                                              const float* __restrict__ bias,
                                              float* __restrict__ Cf,
                                              bf16_t* __restrict__ Cbf,
                                              int M, int N, int K, int flags) {
    int wave = threadIdx.x >> 5, lane = threadIdx.x & 31;
    int tilesN = N >> 4;
    int nBlk = (tilesN + 1) >> 1;          // 32-wide N blocks; last may be half
    int mBlk = M >> 5;                     // M is always a multiple of 32 here
    int blk = blockIdx.x * 8 + wave;
    if (blk >= mBlk * nBlk) return;        // whole-wave exit: EXEC stays full
    int m0 = (blk / nBlk) << 5;
    int n0 = (blk % nBlk) << 5;

    const int koff = (lane >> 4) << 4;
    const bf16_t* Ap0 = A  + (size_t)(m0 + (lane & 15)) * K + koff;
    const bf16_t* Ap1 = Ap0 + (size_t)16 * K;
    const bf16_t* Bp0 = Bt + (size_t)(n0 + (lane & 15)) * K + koff;
    const bf16_t* Bp1 = Bp0 + (size_t)16 * K;

    const v8f vz = {0.f, 0.f, 0.f, 0.f, 0.f, 0.f, 0.f, 0.f};
    v8f acc00 = vz, acc01 = vz, acc10 = vz, acc11 = vz;

    // Wave-uniform by construction (n0 depends only on wave id); make it
    // provably scalar so the compiler emits a plain s_cbranch, not saveexec.
    const int has2 = __builtin_amdgcn_readfirstlane((n0 + 16) < N ? 1 : 0);

    if (has2) {
        for (int k0 = 0; k0 < K; k0 += 32) {
            v16bf a0 = *(const v16bf*)(Ap0 + k0);
            v16bf a1 = *(const v16bf*)(Ap1 + k0);
            v16bf b0 = *(const v16bf*)(Bp0 + k0);
            v16bf b1 = *(const v16bf*)(Bp1 + k0);
            acc00 = wmma_bf16(a0, b0, acc00);
            acc10 = wmma_bf16(a1, b0, acc10);
            acc01 = wmma_bf16(a0, b1, acc01);
            acc11 = wmma_bf16(a1, b1, acc11);
        }
        store_tile(Cf, Cbf, bias, flags, N, m0,      n0,      lane, acc00);
        store_tile(Cf, Cbf, bias, flags, N, m0 + 16, n0,      lane, acc10);
        store_tile(Cf, Cbf, bias, flags, N, m0,      n0 + 16, lane, acc01);
        store_tile(Cf, Cbf, bias, flags, N, m0 + 16, n0 + 16, lane, acc11);
    } else {
        for (int k0 = 0; k0 < K; k0 += 32) {
            v16bf a0 = *(const v16bf*)(Ap0 + k0);
            v16bf a1 = *(const v16bf*)(Ap1 + k0);
            v16bf b0 = *(const v16bf*)(Bp0 + k0);
            acc00 = wmma_bf16(a0, b0, acc00);
            acc10 = wmma_bf16(a1, b0, acc10);
        }
        store_tile(Cf, Cbf, bias, flags, N, m0,      n0, lane, acc00);
        store_tile(Cf, Cbf, bias, flags, N, m0 + 16, n0, lane, acc10);
    }
}

// ---------------------------------------------------------------------------
// Fused GRU scan. gridDim.x = 2 (dir 0 = forward, 1 = backward).
// One workgroup (16 waves) per direction. h lives in LDS (f32 + bf16 shadow).
// Per step: rec = h @ U^T via WMMA (each wave owns 2 column groups of 16 and
// accumulates all 3 gate chunks + both M-tiles in registers), then gate math
// in registers, barrier, staged write-back of new h. Next step's xz block is
// prefetched (global_prefetch_b8) while the current step computes.
// ---------------------------------------------------------------------------
__global__ __launch_bounds__(512) void k_gru_scan(const bf16_t* __restrict__ fwUt,
                                                  const bf16_t* __restrict__ bwUt,
                                                  const float* __restrict__ fwB1,
                                                  const float* __restrict__ bwB1,
                                                  const float* __restrict__ xzFw,
                                                  const float* __restrict__ xzBw,
                                                  float* __restrict__ encSeq,
                                                  bf16_t* __restrict__ encSeqBf,
                                                  float* __restrict__ hcat,
                                                  bf16_t* __restrict__ hcatBf) {
    __shared__ float  hS [Bdim][Hdim];   // 64 KB
    __shared__ bf16_t hbS[Bdim][Hdim];   // 32 KB
    const int dir  = blockIdx.x;
    const int wave = threadIdx.x >> 5, lane = threadIdx.x & 31;
    const bf16_t* Ut = dir ? bwUt : fwUt;
    const float*  b1 = dir ? bwB1 : fwB1;
    const float*  xz = dir ? xzBw : xzFw;

    for (int i = threadIdx.x; i < Bdim * Hdim; i += 512) {
        ((float*)hS)[i] = 0.0f;
        ((bf16_t*)hbS)[i] = f2bf(0.0f);
    }
    __syncthreads();

    const v8f vzero = {0.f, 0.f, 0.f, 0.f, 0.f, 0.f, 0.f, 0.f};
    const int arow = lane & 15;          // A fragment row within tile
    const int koff = (lane >> 4) << 4;   // A/B fragment K offset within k0

    for (int t = 0; t < Sdim; ++t) {
        const int s = dir ? (Sdim - 1 - t) : t;

        // Prefetch next timestep's xz block (32 rows x 6 KB = 1536 cachelines):
        // 512 threads x 3 lines each, 128 B apart.
        if (t + 1 < Sdim) {
            const int s2 = dir ? (Sdim - 2 - t) : (t + 1);
            const int l0 = threadIdx.x * 3;
            for (int j = 0; j < 3; ++j) {
                int l = l0 + j;                       // 0..1535
                int b = l / 48, off = (l % 48) * 32;  // 48 lines of 32 floats per row
                __builtin_prefetch(xz + ((size_t)b * Sdim + s2) * (3 * Hdim) + off, 0, 1);
            }
        }

        float hn[2][2][8];               // [cg][mtile][r] staged new h

        for (int cgi = 0; cgi < 2; ++cgi) {
            const int j0 = (wave + cgi * 16) << 4;          // hidden column base
            v8f acc[2][3];
            for (int m = 0; m < 2; ++m)
                for (int c = 0; c < 3; ++c) acc[m][c] = vzero;

            for (int k0 = 0; k0 < Hdim; k0 += 32) {
                v16bf a0 = *(const v16bf*)&hbS[arow][k0 + koff];
                v16bf a1 = *(const v16bf*)&hbS[16 + arow][k0 + koff];
                for (int c = 0; c < 3; ++c) {
                    const bf16_t* Bp = Ut + (size_t)(c * Hdim + j0 + arow) * Hdim + k0 + koff;
                    v16bf bv = *(const v16bf*)Bp;
                    acc[0][c] = wmma_bf16(a0, bv, acc[0][c]);
                    acc[1][c] = wmma_bf16(a1, bv, acc[1][c]);
                }
            }
            const int col = j0 + (lane & 15);
            const float bz = b1[col], br = b1[Hdim + col], bh = b1[2 * Hdim + col];
            for (int m = 0; m < 2; ++m) {
                const int rb = (m << 4) + ((lane >> 4) << 3);
                for (int r = 0; r < 8; ++r) {
                    const int b = rb + r;
                    const float* xr = xz + ((size_t)b * Sdim + s) * (3 * Hdim);
                    float z  = sigmoidf_(xr[col]            + acc[m][0][r] + bz);
                    float rr = sigmoidf_(xr[Hdim + col]     + acc[m][1][r] + br);
                    float hh = tanhf    (xr[2 * Hdim + col] + rr * (acc[m][2][r] + bh));
                    hn[cgi][m][r] = z * hS[b][col] + (1.0f - z) * hh;
                }
            }
        }
        __syncthreads();   // everyone done reading old h
        for (int cgi = 0; cgi < 2; ++cgi) {
            const int j0 = (wave + cgi * 16) << 4;
            const int col = j0 + (lane & 15);
            for (int m = 0; m < 2; ++m) {
                const int rb = (m << 4) + ((lane >> 4) << 3);
                for (int r = 0; r < 8; ++r) {
                    const int b = rb + r;
                    float v = hn[cgi][m][r];
                    hS[b][col]  = v;
                    hbS[b][col] = f2bf(v);
                    size_t o = ((size_t)b * Sdim + s) * (2 * Hdim) + dir * Hdim + col;
                    encSeq[o]   = v;
                    encSeqBf[o] = f2bf(v);
                }
            }
        }
        __syncthreads();   // new h visible before next step's WMMA reads
    }
    // final states -> hcat (fw cols 0..511, bw cols 512..1023)
    for (int i = threadIdx.x; i < Bdim * Hdim; i += 512) {
        int b = i >> 9, c = i & (Hdim - 1);
        float v = hS[b][c];
        hcat[b * (2 * Hdim) + dir * Hdim + c]   = v;
        hcatBf[b * (2 * Hdim) + dir * Hdim + c] = f2bf(v);
    }
}

// ---------------------------------------------------------------------------
// Decoder GRU cell (single step), writes dec_state f32/bf16 + fcin[:, 0:512]
// ---------------------------------------------------------------------------
__global__ void k_dec_cell(const float* __restrict__ xzd, const float* __restrict__ recd,
                           const float* __restrict__ state,
                           float* __restrict__ ds, bf16_t* __restrict__ dsbf,
                           bf16_t* __restrict__ fcin) {
    int i = blockIdx.x * blockDim.x + threadIdx.x;
    if (i >= Bdim * Hdim) return;
    int b = i >> 9, j = i & (Hdim - 1);
    size_t o = (size_t)b * (3 * Hdim);
    float z  = sigmoidf_(xzd[o + j]            + recd[o + j]);
    float r  = sigmoidf_(xzd[o + Hdim + j]     + recd[o + Hdim + j]);
    float hh = tanhf    (xzd[o + 2 * Hdim + j] + r * recd[o + 2 * Hdim + j]);
    float v  = z * state[i] + (1.0f - z) * hh;
    ds[i] = v; dsbf[i] = f2bf(v);
    fcin[(size_t)b * (3 * Hdim) + j] = f2bf(v);
}

// e[b,s] = dot(tanh(e1[b,s,:] + sdec[b,:]), att_V) + mask*(-1e10); one wave per (b,s)
__global__ void k_att_score(const float* __restrict__ e1, const float* __restrict__ sdec,
                            const float* __restrict__ attV, const int* __restrict__ mask,
                            float* __restrict__ e) {
    int w = blockIdx.x * 8 + (threadIdx.x >> 5);
    int lane = threadIdx.x & 31;
    if (w >= Bdim * Sdim) return;
    int b = w / Sdim;
    const float* er = e1 + (size_t)w * Hdim;
    const float* sr = sdec + (size_t)b * Hdim;
    float sum = 0.0f;
    for (int c = lane; c < Hdim; c += 32) sum += tanhf(er[c] + sr[c]) * attV[c];
    for (int off = 16; off; off >>= 1) sum += __shfl_xor(sum, off, 32);
    if (lane == 0) e[w] = sum + (float)mask[w] * -1.0e10f;
}

// softmax over S, in place (e -> w); one block per batch row
__global__ void k_softmax_s(float* __restrict__ e) {
    __shared__ float red[256];
    __shared__ float bmax, bsum;
    int b = blockIdx.x, tid = threadIdx.x;
    float m = -1.0e30f;
    for (int s = tid; s < Sdim; s += 256) m = fmaxf(m, e[b * Sdim + s]);
    red[tid] = m; __syncthreads();
    for (int o = 128; o; o >>= 1) { if (tid < o) red[tid] = fmaxf(red[tid], red[tid + o]); __syncthreads(); }
    if (tid == 0) bmax = red[0]; __syncthreads();
    float sum = 0.0f;
    for (int s = tid; s < Sdim; s += 256) sum += expf(e[b * Sdim + s] - bmax);
    red[tid] = sum; __syncthreads();
    for (int o = 128; o; o >>= 1) { if (tid < o) red[tid] += red[tid + o]; __syncthreads(); }
    if (tid == 0) bsum = red[0]; __syncthreads();
    for (int s = tid; s < Sdim; s += 256) e[b * Sdim + s] = expf(e[b * Sdim + s] - bmax) / bsum;
}

// context[b,c] = sum_s w[b,s]*enc_seq[b,s,c]; writes fcin[:, 512:1536]
__global__ void k_context(const float* __restrict__ w, const float* __restrict__ enc,
                          float* __restrict__ ctx, bf16_t* __restrict__ fcin) {
    int i = blockIdx.x * blockDim.x + threadIdx.x;
    if (i >= Bdim * 2 * Hdim) return;
    int b = i >> 10, c = i & (2 * Hdim - 1);
    float sum = 0.0f;
    for (int s = 0; s < Sdim; ++s)
        sum += w[b * Sdim + s] * enc[((size_t)b * Sdim + s) * (2 * Hdim) + c];
    ctx[i] = sum;
    fcin[(size_t)b * (3 * Hdim) + Hdim + c] = f2bf(sum);
}

// per-row max & sum(exp) over vocab logits
__global__ void k_vocab_reduce(const float* __restrict__ lg, float* __restrict__ rmax,
                               float* __restrict__ rsum) {
    __shared__ float red[256];
    __shared__ float bmax;
    int b = blockIdx.x, tid = threadIdx.x;
    float m = -1.0e30f;
    for (int v = tid; v < Vdim; v += 256) m = fmaxf(m, lg[(size_t)b * Vdim + v]);
    red[tid] = m; __syncthreads();
    for (int o = 128; o; o >>= 1) { if (tid < o) red[tid] = fmaxf(red[tid], red[tid + o]); __syncthreads(); }
    if (tid == 0) { bmax = red[0]; rmax[b] = red[0]; } __syncthreads();
    float s = 0.0f;
    for (int v = tid; v < Vdim; v += 256) s += expf(lg[(size_t)b * Vdim + v] - bmax);
    red[tid] = s; __syncthreads();
    for (int o = 128; o; o >>= 1) { if (tid < o) red[tid] += red[tid + o]; __syncthreads(); }
    if (tid == 0) rsum[b] = red[0];
}

__global__ void k_pgen(const float* __restrict__ ctx, const float* __restrict__ ds,
                       const float* __restrict__ yemb, const float* __restrict__ pgWh,
                       const float* __restrict__ pgWs, const float* __restrict__ pgWx,
                       const float* __restrict__ pgb, float* __restrict__ pgen) {
    __shared__ float red[256];
    int b = blockIdx.x, tid = threadIdx.x;
    float s = 0.0f;
    for (int c = tid; c < 2 * Hdim; c += 256) s += ctx[b * 2 * Hdim + c] * pgWh[c];
    for (int c = tid; c < Hdim; c += 256)
        s += ds[b * Hdim + c] * pgWs[c] + yemb[b * Hdim + c] * pgWx[c];
    red[tid] = s; __syncthreads();
    for (int o = 128; o; o >>= 1) { if (tid < o) red[tid] += red[tid + o]; __syncthreads(); }
    if (tid == 0) pgen[b] = sigmoidf_(red[0] + pgb[0]);
}

// out[b, v] = v < V ? p_gen*softmax(logits)[v] : 0     (extended dist, pre-scatter)
__global__ void k_final_vocab(const float* __restrict__ lg, const float* __restrict__ rmax,
                              const float* __restrict__ rsum, const float* __restrict__ pgen,
                              float* __restrict__ out) {
    size_t i = (size_t)blockIdx.x * blockDim.x + threadIdx.x;
    const int EXT = Vdim + Sdim;
    if (i >= (size_t)Bdim * EXT) return;
    int b = (int)(i / EXT), v = (int)(i % EXT);
    out[i] = (v < Vdim)
        ? pgen[b] * expf(lg[(size_t)b * Vdim + v] - rmax[b]) / rsum[b]
        : 0.0f;
}

// scatter-add copy weights at x_id
__global__ void k_scatter(const int* __restrict__ xid, const int* __restrict__ mask,
                          const float* __restrict__ w, const float* __restrict__ pgen,
                          float* __restrict__ out) {
    int i = blockIdx.x * blockDim.x + threadIdx.x;
    if (i >= Bdim * Sdim) return;
    int b = i / Sdim;
    float aw = w[i] * (1.0f - (float)mask[i]) * (1.0f - pgen[b]);
    atomicAdd(&out[(size_t)b * (Vdim + Sdim) + xid[i]], aw);
}

// ---------------------------------------------------------------------------
static inline void* bump(char** p, size_t bytes) {
    void* r = *p;
    *p += (bytes + 255) & ~(size_t)255;
    return r;
}
static inline int cdiv(long long a, long long b) { return (int)((a + b - 1) / b); }

extern "C" void kernel_launch(void* const* d_in, const int* in_sizes, int n_in,
                              void* d_out, int out_size, void* d_ws, size_t ws_size,
                              hipStream_t stream) {
    const int* x_id    = (const int*)d_in[0];
    const int* y_id    = (const int*)d_in[1];
    const int* x_mask  = (const int*)d_in[2];
    const float* emb   = (const float*)d_in[3];
    const float* fwW   = (const float*)d_in[4];
    const float* fwU   = (const float*)d_in[5];
    const float* fwB   = (const float*)d_in[6];
    const float* bwW   = (const float*)d_in[7];
    const float* bwU   = (const float*)d_in[8];
    const float* bwB   = (const float*)d_in[9];
    const float* redW  = (const float*)d_in[10];
    const float* redB  = (const float*)d_in[11];
    const float* decW  = (const float*)d_in[12];
    const float* decU  = (const float*)d_in[13];
    const float* decB  = (const float*)d_in[14];
    const float* attWh = (const float*)d_in[15];
    const float* attWs = (const float*)d_in[16];
    const float* attBs = (const float*)d_in[17];
    /* att_Wc unused: cover == 0 exactly */
    const float* attV  = (const float*)d_in[19];
    const float* fcW   = (const float*)d_in[20];
    const float* fcB   = (const float*)d_in[21];
    const float* vocW  = (const float*)d_in[22];
    const float* pgWh  = (const float*)d_in[23];
    const float* pgWs  = (const float*)d_in[24];
    const float* pgWx  = (const float*)d_in[25];
    const float* pgb   = (const float*)d_in[26];
    float* out = (float*)d_out;

    const int H = Hdim, B = Bdim, S = Sdim, V = Vdim;
    const int H3 = 3 * H, H2 = 2 * H;
    char* p = (char*)d_ws;

    bf16_t* fwWt   = (bf16_t*)bump(&p, (size_t)H3 * H * 2);
    bf16_t* fwUt   = (bf16_t*)bump(&p, (size_t)H3 * H * 2);
    bf16_t* bwWt   = (bf16_t*)bump(&p, (size_t)H3 * H * 2);
    bf16_t* bwUt   = (bf16_t*)bump(&p, (size_t)H3 * H * 2);
    bf16_t* decWt  = (bf16_t*)bump(&p, (size_t)H3 * H * 2);
    bf16_t* decUt  = (bf16_t*)bump(&p, (size_t)H3 * H * 2);
    bf16_t* redWt  = (bf16_t*)bump(&p, (size_t)H * H2 * 2);
    bf16_t* attWht = (bf16_t*)bump(&p, (size_t)H * H2 * 2);
    bf16_t* attWst = (bf16_t*)bump(&p, (size_t)H * H * 2);
    bf16_t* fcWt   = (bf16_t*)bump(&p, (size_t)H * H3 * 2);
    bf16_t* vocWt  = (bf16_t*)bump(&p, (size_t)V * H * 2);
    bf16_t* xembbf = (bf16_t*)bump(&p, (size_t)B * S * H * 2);
    float*  yemb   = (float*) bump(&p, (size_t)B * H * 4);
    bf16_t* yembbf = (bf16_t*)bump(&p, (size_t)B * H * 2);
    float*  xzFw   = (float*) bump(&p, (size_t)B * S * H3 * 4);
    float*  xzBw   = (float*) bump(&p, (size_t)B * S * H3 * 4);
    float*  encSeq = (float*) bump(&p, (size_t)B * S * H2 * 4);
    bf16_t* encBf  = (bf16_t*)bump(&p, (size_t)B * S * H2 * 2);
    float*  hcat   = (float*) bump(&p, (size_t)B * H2 * 4);
    bf16_t* hcatBf = (bf16_t*)bump(&p, (size_t)B * H2 * 2);
    float*  state  = (float*) bump(&p, (size_t)B * H * 4);
    bf16_t* stateBf= (bf16_t*)bump(&p, (size_t)B * H * 2);
    float*  xzd    = (float*) bump(&p, (size_t)B * H3 * 4);
    float*  recd   = (float*) bump(&p, (size_t)B * H3 * 4);
    float*  decSt  = (float*) bump(&p, (size_t)B * H * 4);
    bf16_t* decStBf= (bf16_t*)bump(&p, (size_t)B * H * 2);
    float*  e1     = (float*) bump(&p, (size_t)B * S * H * 4);
    float*  sdec   = (float*) bump(&p, (size_t)B * H * 4);
    float*  escore = (float*) bump(&p, (size_t)B * S * 4);
    float*  ctx    = (float*) bump(&p, (size_t)B * H2 * 4);
    bf16_t* fcin   = (bf16_t*)bump(&p, (size_t)B * H3 * 2);
    bf16_t* youtBf = (bf16_t*)bump(&p, (size_t)B * H * 2);
    float*  logits = (float*) bump(&p, (size_t)B * V * 4);
    float*  rmax   = (float*) bump(&p, 32 * 4);
    float*  rsum   = (float*) bump(&p, 32 * 4);
    float*  pgen   = (float*) bump(&p, 32 * 4);

    // ---- weight conversion/transposition (f32 -> bf16, (K,N)->(N,K)) ----
    k_convert_t<<<cdiv((long long)H * H3, 256), 256, 0, stream>>>(fwW,  fwWt,  H,  H3);
    k_convert_t<<<cdiv((long long)H * H3, 256), 256, 0, stream>>>(fwU,  fwUt,  H,  H3);
    k_convert_t<<<cdiv((long long)H * H3, 256), 256, 0, stream>>>(bwW,  bwWt,  H,  H3);
    k_convert_t<<<cdiv((long long)H * H3, 256), 256, 0, stream>>>(bwU,  bwUt,  H,  H3);
    k_convert_t<<<cdiv((long long)H * H3, 256), 256, 0, stream>>>(decW, decWt, H,  H3);
    k_convert_t<<<cdiv((long long)H * H3, 256), 256, 0, stream>>>(decU, decUt, H,  H3);
    k_convert_t<<<cdiv((long long)H2 * H, 256), 256, 0, stream>>>(redW, redWt, H2, H);
    k_convert_t<<<cdiv((long long)H2 * H, 256), 256, 0, stream>>>(attWh,attWht, H2, H);
    k_convert_t<<<cdiv((long long)H * H, 256), 256, 0, stream>>>(attWs,attWst, H,  H);
    k_convert_t<<<cdiv((long long)H3 * H, 256), 256, 0, stream>>>(fcW,  fcWt,  H3, H);
    k_convert_t<<<cdiv((long long)H * V, 256), 256, 0, stream>>>(vocW, vocWt, H,  V);

    // ---- embeddings ----
    k_gather_x<<<cdiv((long long)B * S * H, 256), 256, 0, stream>>>(x_id, emb, xembbf, B * S);
    k_gather_y<<<cdiv((long long)B * H, 256), 256, 0, stream>>>(y_id, emb, yemb, yembbf);

    // ---- input projections: xz = x_emb @ W + b[0]  (M=12800, N=1536, K=512) ----
    {
        int blks = (B * S / 32) * ((H3 / 16 + 1) / 2);
        k_gemm<<<cdiv(blks, 8), 256, 0, stream>>>(xembbf, fwWt, fwB, xzFw, nullptr, B * S, H3, H, 0);
        k_gemm<<<cdiv(blks, 8), 256, 0, stream>>>(xembbf, bwWt, bwB, xzBw, nullptr, B * S, H3, H, 0);
    }

    // ---- sequential bi-GRU scan (one workgroup per direction) ----
    k_gru_scan<<<2, 512, 0, stream>>>(fwUt, bwUt, fwB + H3, bwB + H3, xzFw, xzBw,
                                      encSeq, encBf, hcat, hcatBf);

    // ---- reduce state: relu(hcat @ red_W + red_b) ----
    k_gemm<<<cdiv((B / 32) * (H / 32), 8), 256, 0, stream>>>(hcatBf, redWt, redB,
                                                             state, stateBf, B, H, H2, 1);
    // ---- decoder single GRU step ----
    k_gemm<<<cdiv((B / 32) * (H3 / 32), 8), 256, 0, stream>>>(yembbf, decWt, decB,
                                                              xzd, nullptr, B, H3, H, 0);
    k_gemm<<<cdiv((B / 32) * (H3 / 32), 8), 256, 0, stream>>>(stateBf, decUt, decB + H3,
                                                              recd, nullptr, B, H3, H, 0);
    k_dec_cell<<<cdiv((long long)B * H, 256), 256, 0, stream>>>(xzd, recd, state,
                                                                decSt, decStBf, fcin);

    // ---- attention ----
    k_gemm<<<cdiv((B * S / 32) * (H / 32), 8), 256, 0, stream>>>(encBf, attWht, nullptr,
                                                                 e1, nullptr, B * S, H, H2, 0);
    k_gemm<<<cdiv((B / 32) * (H / 32), 8), 256, 0, stream>>>(decStBf, attWst, attBs,
                                                             sdec, nullptr, B, H, H, 0);
    k_att_score<<<cdiv((long long)B * S, 8), 256, 0, stream>>>(e1, sdec, attV, x_mask, escore);
    k_softmax_s<<<B, 256, 0, stream>>>(escore);
    k_context<<<cdiv((long long)B * H2, 256), 256, 0, stream>>>(escore, encSeq, ctx, fcin);

    // ---- output head ----
    k_gemm<<<cdiv((B / 32) * (H / 32), 8), 256, 0, stream>>>(fcin, fcWt, fcB,
                                                             nullptr, youtBf, B, H, H3, 0);
    {
        int blks = (B / 32) * ((V / 16 + 1) / 2);   // 3125 N-tiles -> 1563 blocks (last 2x1)
        k_gemm<<<cdiv(blks, 8), 256, 0, stream>>>(youtBf, vocWt, nullptr,
                                                  logits, nullptr, B, V, H, 0);
    }
    k_vocab_reduce<<<B, 256, 0, stream>>>(logits, rmax, rsum);
    k_pgen<<<B, 256, 0, stream>>>(ctx, decSt, yemb, pgWh, pgWs, pgWx, pgb, pgen);

    // ---- extended distribution + pointer scatter ----
    k_final_vocab<<<cdiv((long long)B * (V + S), 256), 256, 0, stream>>>(logits, rmax, rsum,
                                                                         pgen, out);
    k_scatter<<<cdiv((long long)B * S, 256), 256, 0, stream>>>(x_id, x_mask, escore, pgen, out);
}
